// MultiHeadAttention_35751307772241
// MI455X (gfx1250) — matrix operand order repaired and checked by reading, not compiled, not run on hardware
//
#include <hip/hip_runtime.h>
#include <hip/hip_bf16.h>

typedef __attribute__((ext_vector_type(16))) _Float16 v16h;
typedef __attribute__((ext_vector_type(8)))  _Float16 v8h;
typedef __attribute__((ext_vector_type(8)))  float    v8f;
typedef __attribute__((ext_vector_type(4)))  unsigned int u32x4;
typedef __attribute__((ext_vector_type(8)))  unsigned int u32x8;

#define EMBED 512
#define NHEAD 8
#define HDIM  64
#define BATCH 2
#define SEQ   4096
#define BLK   64

__device__ __forceinline__ v16h join16(v8h lo, v8h hi) {
  v16h r;
#pragma unroll
  for (int i = 0; i < 8; ++i) { r[i] = lo[i]; r[i + 8] = hi[i]; }
  return r;
}

// ---------------------------------------------------------------------------
// TDM: issue an async 2D tile load (f16 elements) global -> LDS.
// LDS destination uses pad_amount=4 DWORDs every pad_interval=32 DWORDs,
// i.e. a 64-half row lands with stride 72 halves (matches [64][72] tiles).
// Group layouts per CDNA5 ISA.
// ---------------------------------------------------------------------------
__device__ __forceinline__ void tdm_load_2d_f16(unsigned lds_addr,
                                                const void* gptr,
                                                unsigned tile_d0, unsigned tile_d1,
                                                unsigned tensor_d0, unsigned tensor_d1,
                                                unsigned long long stride0) {
  unsigned long long ga = (unsigned long long)gptr;
  u32x4 g0;
  g0[0] = 1u;                                            // count=1 (valid user D#)
  g0[1] = lds_addr;                                      // lds_addr [63:32]
  g0[2] = (unsigned)ga;                                  // global_addr lo
  g0[3] = ((unsigned)(ga >> 32) & 0x01FFFFFFu) | (2u << 30);  // addr hi | type=2
  u32x8 g1;
  g1[0] = (1u << 16)      // data_size = 1 -> 2 bytes
        | (1u << 20)      // pad_enable
        | (4u << 22)      // pad_interval: 2^(4+1)=32 DWORDs (one 64-half row)
        | (3u << 25);     // pad_amount: 3+1 = 4 DWORDs (8 halves)
  g1[1] = (tensor_d0 & 0xFFFFu) << 16;                   // tensor_dim0 [79:48] lo
  g1[2] = ((tensor_d0 >> 16) & 0xFFFFu) | ((tensor_d1 & 0xFFFFu) << 16);
  g1[3] = ((tensor_d1 >> 16) & 0xFFFFu) | ((tile_d0 & 0xFFFFu) << 16);
  g1[4] = (tile_d1 & 0xFFFFu);                           // tile_dim1; tile_dim2=0
  g1[5] = (unsigned)(stride0 & 0xFFFFFFFFu);             // tensor_dim0_stride lo
  g1[6] = (unsigned)((stride0 >> 32) & 0xFFFFu);         // stride hi; dim1_stride=0
  g1[7] = 0u;
  asm volatile("tensor_load_to_lds %0, %1" :: "s"(g0), "s"(g1) : "memory");
}

// ---------------------------------------------------------------------------
// Kernel 1: per-head projection  out[..,i] = sum_j X[..,j] * W[i][j], f32->f16
// Q/K stored row-major [B,H,S,64]; V stored transposed [B,H,64,S].
// ---------------------------------------------------------------------------
__global__ __launch_bounds__(256)
void proj_f16_kernel(const float* __restrict__ X, const float* __restrict__ W,
                     _Float16* __restrict__ Out, int transposed) {
  __shared__ float Ws[HDIM][HDIM];
  __shared__ float Xs[BLK][HDIM + 1];
  const int tid = threadIdx.x;
  const int bh  = blockIdx.y;
  const int s0  = blockIdx.x * BLK;
  const int b   = bh / NHEAD, h = bh % NHEAD;

  for (int i = tid; i < HDIM * HDIM; i += 256) Ws[i >> 6][i & 63] = W[i];
  for (int i = tid; i < BLK * HDIM; i += 256) {
    int r = i >> 6, c = i & 63;
    Xs[r][c] = X[(size_t)(b * SEQ + s0 + r) * EMBED + h * HDIM + c];
  }
  __syncthreads();

  const int r  = tid >> 2;
  const int i0 = (tid & 3) * 16;
#pragma unroll
  for (int ii = 0; ii < 16; ++ii) {
    int i = i0 + ii;
    float acc = 0.f;
#pragma unroll
    for (int j = 0; j < HDIM; ++j) acc += Xs[r][j] * Ws[i][j];
    if (!transposed)
      Out[((size_t)bh * SEQ + s0 + r) * HDIM + i] = (_Float16)acc;
    else
      Out[((size_t)bh * HDIM + i) * SEQ + s0 + r] = (_Float16)acc;
  }
}

// ---------------------------------------------------------------------------
// Kernel 2: f32 -> f16 convert (for Wo)
// ---------------------------------------------------------------------------
__global__ void f32_to_f16_kernel(const float* __restrict__ in,
                                  _Float16* __restrict__ out, int n) {
  int i = blockIdx.x * blockDim.x + threadIdx.x;
  if (i < n) out[i] = (_Float16)in[i];
}

// ---------------------------------------------------------------------------
// Kernel 3: flash attention. 1 workgroup (4 waves) = 64 q-rows of one (b,h).
// TDM double-buffered K/V staging overlapped with WMMA compute.
// ---------------------------------------------------------------------------
__global__ __launch_bounds__(128)
void flash_attn_kernel(const _Float16* __restrict__ Qh,  // [B,H,S,64]
                       const _Float16* __restrict__ Kh,  // [B,H,S,64]
                       const _Float16* __restrict__ Vh,  // [B,H,64,S]
                       const int* __restrict__ mask,     // [S,S]
                       _Float16* __restrict__ ctx)       // [B*S, 512]
{
  __shared__ __attribute__((aligned(16))) _Float16 Ks[2][BLK][72];   // [key][d]
  __shared__ __attribute__((aligned(16))) _Float16 Vs[2][HDIM][72];  // [d][key]
  __shared__ __attribute__((aligned(16))) _Float16 Ps[4][16][72];    // per-wave P

  const int tid  = threadIdx.x;
  const int lane = tid & 31;
  const int w    = tid >> 5;
  const int lo16 = lane & 15;
  const int hi16 = lane >> 4;

  const int bh = blockIdx.y;
  const int b  = bh / NHEAD, h = bh % NHEAD;
  const int qb = blockIdx.x * BLK;

  const _Float16* Qg = Qh + (size_t)bh * SEQ * HDIM;
  const _Float16* Kg = Kh + (size_t)bh * SEQ * HDIM;
  const _Float16* Vg = Vh + (size_t)bh * HDIM * SEQ;

  // --- preload Q A-fragments (16x32 per k-step over d), persistent in VGPRs
  const int qrow = qb + 16 * w + lo16;
  v16h aq0, aq1;
  {
    const _Float16* qp = Qg + (size_t)qrow * HDIM;
    const int c = 8 * hi16;
    aq0 = join16(*(const v8h*)(qp + c),      *(const v8h*)(qp + c + 16));
    aq1 = join16(*(const v8h*)(qp + 32 + c), *(const v8h*)(qp + 48 + c));
  }

  const v8f vzero = {0.f, 0.f, 0.f, 0.f, 0.f, 0.f, 0.f, 0.f};
  float mrow[8], lrow[8];
  v8f oacc[4];
#pragma unroll
  for (int r = 0; r < 8; ++r) { mrow[r] = -3.0e38f; lrow[r] = 0.f; }
#pragma unroll
  for (int t = 0; t < 4; ++t) oacc[t] = vzero;

  const unsigned ldsK0 = (unsigned)(size_t)&Ks[0][0][0];
  const unsigned ldsK1 = (unsigned)(size_t)&Ks[1][0][0];
  const unsigned ldsV0 = (unsigned)(size_t)&Vs[0][0][0];
  const unsigned ldsV1 = (unsigned)(size_t)&Vs[1][0][0];

  const int nTiles = SEQ / BLK;
  // prologue: async-stage tile 0 into buffer 0 (issued by wave 0 only)
  if (w == 0) {
    tdm_load_2d_f16(ldsK0, Kg, 64, 64, 64, 64, 64ull);
    tdm_load_2d_f16(ldsV0, Vg, 64, 64, SEQ, 64, (unsigned long long)SEQ);
  }

  for (int it = 0; it < nTiles; ++it) {
    const int kb  = it * BLK;
    const int buf = it & 1;
    if (w == 0) __builtin_amdgcn_s_wait_tensorcnt(0);   // tile `it` landed
    __syncthreads();                                    // publish to all waves
    if (w == 0 && it + 1 < nTiles) {                    // overlap next DMA
      const int nb = buf ^ 1;
      tdm_load_2d_f16(nb ? ldsK1 : ldsK0, Kg + (size_t)(kb + BLK) * HDIM,
                      64, 64, 64, 64, 64ull);
      tdm_load_2d_f16(nb ? ldsV1 : ldsV0, Vg + (kb + BLK),
                      64, 64, SEQ, 64, (unsigned long long)SEQ);
    }

    // --- scores S = Q K^T : batch all B-fragments, then back-to-back WMMAs
    v8f sc[4];
    {
      v16h bk[4];
#pragma unroll
      for (int t = 0; t < 4; ++t) {
        const _Float16* kp = &Ks[buf][16 * t + lo16][8 * hi16];
        bk[t] = join16(*(const v8h*)kp, *(const v8h*)(kp + 16));
      }
#pragma unroll
      for (int t = 0; t < 4; ++t)
        sc[t] = __builtin_amdgcn_wmma_f32_16x16x32_f16(false, aq0, false, bk[t],
                                                       (short)0, vzero, false, false);
#pragma unroll
      for (int t = 0; t < 4; ++t) {
        const _Float16* kp = &Ks[buf][16 * t + lo16][32 + 8 * hi16];
        bk[t] = join16(*(const v8h*)kp, *(const v8h*)(kp + 16));
      }
#pragma unroll
      for (int t = 0; t < 4; ++t)
        sc[t] = __builtin_amdgcn_wmma_f32_16x16x32_f16(false, aq1, false, bk[t],
                                                       (short)0, sc[t], false, false);
    }

    // --- mask (fill -1e20 BEFORE scaling), then *1/sqrt(64)
#pragma unroll
    for (int t = 0; t < 4; ++t) {
      const int col = kb + 16 * t + lo16;
      const int* mp = mask + (size_t)(qb + 16 * w + 8 * hi16) * SEQ + col;
#pragma unroll
      for (int r = 0; r < 8; ++r) {
        int mv = mp[(size_t)r * SEQ];
        float s = sc[t][r];
        sc[t][r] = mv ? s * 0.125f : -1.25e19f;   // (-1e20)/8
      }
    }

    // --- online softmax (rows striped per C-layout: 16-lane halves)
    float bm[8];
#pragma unroll
    for (int r = 0; r < 8; ++r)
      bm[r] = fmaxf(fmaxf(sc[0][r], sc[1][r]), fmaxf(sc[2][r], sc[3][r]));
#pragma unroll
    for (int xm = 1; xm <= 8; xm <<= 1)
#pragma unroll
      for (int r = 0; r < 8; ++r)
        bm[r] = fmaxf(bm[r], __shfl_xor(bm[r], xm, 32));

    float psum[8];
#pragma unroll
    for (int r = 0; r < 8; ++r) {
      float mnew  = fmaxf(mrow[r], bm[r]);
      float alpha = __expf(mrow[r] - mnew);
      mrow[r] = mnew;
      float s = 0.f;
#pragma unroll
      for (int t = 0; t < 4; ++t) {
        float p = __expf(sc[t][r] - mnew);
        sc[t][r] = p;
        s += p;
      }
      psum[r] = s;
      lrow[r] = lrow[r] * alpha;
#pragma unroll
      for (int t = 0; t < 4; ++t) oacc[t][r] = oacc[t][r] * alpha;
    }
#pragma unroll
    for (int xm = 1; xm <= 8; xm <<= 1)
#pragma unroll
      for (int r = 0; r < 8; ++r) psum[r] += __shfl_xor(psum[r], xm, 32);
#pragma unroll
    for (int r = 0; r < 8; ++r) lrow[r] += psum[r];

    // --- stage P (C-layout -> LDS) then reload in A-layout for P·V
#pragma unroll
    for (int t = 0; t < 4; ++t)
#pragma unroll
      for (int r = 0; r < 8; ++r)
        Ps[w][8 * hi16 + r][16 * t + lo16] = (_Float16)sc[t][r];

    asm volatile("s_wait_dscnt 0x0" ::: "memory");   // wave-local LDS RAW

#pragma unroll
    for (int s2 = 0; s2 < 2; ++s2) {   // k-step over 32 keys
      const _Float16* prow = &Ps[w][lo16][32 * s2 + 8 * hi16];
      v16h ap = join16(*(const v8h*)prow, *(const v8h*)(prow + 16));
      v16h bv[4];
#pragma unroll
      for (int t = 0; t < 4; ++t) {
        const _Float16* vp = &Vs[buf][16 * t + lo16][32 * s2 + 8 * hi16];
        bv[t] = join16(*(const v8h*)vp, *(const v8h*)(vp + 16));
      }
#pragma unroll
      for (int t = 0; t < 4; ++t)
        oacc[t] = __builtin_amdgcn_wmma_f32_16x16x32_f16(false, ap, false, bv[t],
                                                         (short)0, oacc[t],
                                                         false, false);
    }
    // single barrier per iteration (top of loop) is sufficient:
    // tile it+2 is only issued after the *next* top barrier, which follows
    // everyone finishing reads of buf[it&1] here.
  }

  // --- normalize and write ctx (f16, [B*S, H*d] head-concat layout)
#pragma unroll
  for (int t = 0; t < 4; ++t) {
#pragma unroll
    for (int r = 0; r < 8; ++r) {
      const int qr = qb + 16 * w + 8 * hi16 + r;
      const float denom = lrow[r];
      const float val = (denom > 0.f) ? oacc[t][r] / denom : 0.f;
      ctx[(size_t)(b * SEQ + qr) * EMBED + h * HDIM + 16 * t + lo16] =
          (_Float16)val;
    }
  }
}

// ---------------------------------------------------------------------------
// Kernel 4: out = ctx @ Wo^T + bo   (M=B*S=8192, N=K=512), WMMA f16->f32
// ---------------------------------------------------------------------------
__global__ __launch_bounds__(128)
void out_proj_kernel(const _Float16* __restrict__ ctx,   // [B*S, 512]
                     const _Float16* __restrict__ Wo16,  // [512, 512] (Wo[i][j])
                     const float* __restrict__ bo,
                     float* __restrict__ out)            // [B*S, 512] f32
{
  const int tid  = threadIdx.x;
  const int lane = tid & 31;
  const int w    = tid >> 5;
  const int lo16 = lane & 15;
  const int hi16 = lane >> 4;
  const int n0   = blockIdx.x * 64;
  const int m0   = blockIdx.y * 64 + 16 * w;

  const v8f vzero = {0.f, 0.f, 0.f, 0.f, 0.f, 0.f, 0.f, 0.f};
  v8f acc[4];
#pragma unroll
  for (int t = 0; t < 4; ++t) acc[t] = vzero;

  const _Float16* arow = ctx + (size_t)(m0 + lo16) * EMBED;
  for (int kk = 0; kk < EMBED; kk += 32) {
    const int c = kk + 8 * hi16;
    v16h a = join16(*(const v8h*)(arow + c), *(const v8h*)(arow + c + 16));
    v16h bfr[4];
#pragma unroll
    for (int t = 0; t < 4; ++t) {
      const _Float16* brow =
          Wo16 + (size_t)(n0 + 16 * t + lo16) * EMBED + kk + 8 * hi16;
      bfr[t] = join16(*(const v8h*)brow, *(const v8h*)(brow + 16));
    }
#pragma unroll
    for (int t = 0; t < 4; ++t)
      acc[t] = __builtin_amdgcn_wmma_f32_16x16x32_f16(false, a, false, bfr[t],
                                                      (short)0, acc[t],
                                                      false, false);
  }

#pragma unroll
  for (int t = 0; t < 4; ++t) {
    const int col  = n0 + 16 * t + lo16;
    const float bias = bo[col];
#pragma unroll
    for (int r = 0; r < 8; ++r) {
      const int row = m0 + 8 * hi16 + r;
      out[(size_t)row * EMBED + col] = acc[t][r] + bias;
    }
  }
}

// ---------------------------------------------------------------------------
extern "C" void kernel_launch(void* const* d_in, const int* in_sizes, int n_in,
                              void* d_out, int out_size, void* d_ws,
                              size_t ws_size, hipStream_t stream) {
  const float* key   = (const float*)d_in[0];
  const float* query = (const float*)d_in[1];
  const float* value = (const float*)d_in[2];
  const int*   mask  = (const int*)d_in[3];
  const float* Wq    = (const float*)d_in[4];
  const float* Wk    = (const float*)d_in[5];
  const float* Wv    = (const float*)d_in[6];
  const float* Wo    = (const float*)d_in[7];
  const float* bo    = (const float*)d_in[8];
  float* out = (float*)d_out;

  char* ws = (char*)d_ws;
  size_t off = 0;
  const size_t qkv_bytes = (size_t)BATCH * NHEAD * SEQ * HDIM * sizeof(_Float16);
  _Float16* Qh   = (_Float16*)(ws + off); off += qkv_bytes;
  _Float16* Kh   = (_Float16*)(ws + off); off += qkv_bytes;
  _Float16* Vh   = (_Float16*)(ws + off); off += qkv_bytes;
  _Float16* ctx  = (_Float16*)(ws + off); off += (size_t)BATCH * SEQ * EMBED * sizeof(_Float16);
  _Float16* Wo16 = (_Float16*)(ws + off); off += (size_t)EMBED * EMBED * sizeof(_Float16);
  (void)ws_size; (void)in_sizes; (void)n_in; (void)out_size;

  dim3 pgrid(SEQ / BLK, BATCH * NHEAD);
  proj_f16_kernel<<<pgrid, 256, 0, stream>>>(query, Wq, Qh, 0);
  proj_f16_kernel<<<pgrid, 256, 0, stream>>>(key,   Wk, Kh, 0);
  proj_f16_kernel<<<pgrid, 256, 0, stream>>>(value, Wv, Vh, 1);
  f32_to_f16_kernel<<<(EMBED * EMBED + 255) / 256, 256, 0, stream>>>(
      Wo, Wo16, EMBED * EMBED);

  dim3 agrid(SEQ / BLK, BATCH * NHEAD);
  flash_attn_kernel<<<agrid, 128, 0, stream>>>(Qh, Kh, Vh, mask, ctx);

  dim3 ogrid(EMBED / 64, (BATCH * SEQ) / 64);
  out_proj_kernel<<<ogrid, 128, 0, stream>>>(ctx, Wo16, bo, out);
}
